// Model_1958505087697
// MI455X (gfx1250) — compile-verified, matrix operand order checked by reading, not run on hardware
//
#include <hip/hip_runtime.h>
#include <math.h>

typedef __attribute__((ext_vector_type(16))) _Float16 v16h;
typedef __attribute__((ext_vector_type(8)))  float    v8f;

// Problem constants (match reference setup_inputs)
#define Bb   4
#define Nn   128
#define Mm   16
#define Kk   80
#define CF   256
#define TF   32
#define CS   2048
#define TS   8
#define Hh   14
#define Ww   14
#define Cc   (CF + CS)        // 2304
#define HW   (Hh * Ww)        // 196
#define PP   7
#define FG_HI 0.85f
#define FG_LO 0.5f
#define STRIDE_INV (1.0f / 16.0f)

// GEMM tiling: logits[512,80] = hid[512,2304] x W[2304,80]
#define MTILES 32             // 512/16
#define NTILES 5              // 80/16
#define KTILES 72             // 2304/32

// ---------------------------------------------------------------------------
// Kernel 1: temporal mean over T + channel concat -> feature [B, C, H, W] f32
// ---------------------------------------------------------------------------
__global__ void k_temporal_mean(const float* __restrict__ fast,
                                const float* __restrict__ slow,
                                float* __restrict__ feature) {
    const int total = Bb * Cc * HW;
    for (int idx = blockIdx.x * blockDim.x + threadIdx.x; idx < total;
         idx += gridDim.x * blockDim.x) {
        int hw = idx % HW;
        int c  = (idx / HW) % Cc;
        int b  = idx / (HW * Cc);
        float acc = 0.0f;
        if (c < CF) {
            const float* p = fast + ((size_t)(b * CF + c) * TF) * HW + hw;
            #pragma unroll
            for (int t = 0; t < TF; ++t) acc += p[t * HW];
            acc *= (1.0f / TF);
        } else {
            const float* p = slow + ((size_t)(b * CS + (c - CF)) * TS) * HW + hw;
            #pragma unroll
            for (int t = 0; t < TS; ++t) acc += p[t * HW];
            acc *= (1.0f / TS);
        }
        feature[idx] = acc;
    }
}

// ---------------------------------------------------------------------------
// Kernel 2: IoU assignment + fg mask (global any() fallback) + ROI grid idx.
// One block of B*N = 512 threads so the global fg_hi.any() is a block reduce.
// ---------------------------------------------------------------------------
__global__ void k_assign(const float* __restrict__ props,   // [B,N,4]
                         const float* __restrict__ gt,      // [B,M,4]
                         int*   __restrict__ assign,        // [B*N]
                         float* __restrict__ fg,            // [B*N]
                         int*   __restrict__ iy,            // [B*N,7]
                         int*   __restrict__ ix) {          // [B*N,7]
    __shared__ int s_any;
    if (threadIdx.x == 0) s_any = 0;
    __syncthreads();

    const int t = threadIdx.x;           // 0..511
    const int b = t / Nn;
    const int n = t % Nn;
    const float* pb = props + (size_t)(b * Nn + n) * 4;
    const float ax1 = pb[0], ay1 = pb[1], ax2 = pb[2], ay2 = pb[3];
    const float area_a = (ax2 - ax1) * (ay2 - ay1);

    float best = -1.0f;
    int   barg = 0;
    for (int m = 0; m < Mm; ++m) {
        const float* gb = gt + (size_t)(b * Mm + m) * 4;
        const float gx1 = gb[0], gy1 = gb[1], gx2 = gb[2], gy2 = gb[3];
        const float tlx = fmaxf(ax1, gx1), tly = fmaxf(ay1, gy1);
        const float brx = fminf(ax2, gx2), bry = fminf(ay2, gy2);
        const float iw = fmaxf(brx - tlx, 0.0f), ih = fmaxf(bry - tly, 0.0f);
        const float inter  = iw * ih;
        const float area_g = (gx2 - gx1) * (gy2 - gy1);
        const float iou = inter / (area_a + area_g - inter);
        if (iou > best) { best = iou; barg = m; }
    }
    const int fg_hi = (best >= FG_HI) ? 1 : 0;
    if (fg_hi) atomicOr(&s_any, 1);
    __syncthreads();
    const int any_hi = s_any;

    assign[t] = barg;
    fg[t] = any_hi ? (float)fg_hi : ((best >= FG_LO) ? 1.0f : 0.0f);

    // ROI grid indices (nearest-floor sampling on the 14x14 feature map)
    const float x1 = ax1 * STRIDE_INV, y1 = ay1 * STRIDE_INV;
    const float x2 = ax2 * STRIDE_INV, y2 = ay2 * STRIDE_INV;
    #pragma unroll
    for (int p = 0; p < PP; ++p) {
        const float g  = ((float)p + 0.5f) / (float)PP;
        const float cy = y1 + g * (y2 - y1);
        const float cx = x1 + g * (x2 - x1);
        int yi = (int)floorf(cy); yi = yi < 0 ? 0 : (yi > Hh - 1 ? Hh - 1 : yi);
        int xi = (int)floorf(cx); xi = xi < 0 ? 0 : (xi > Ww - 1 ? Ww - 1 : xi);
        iy[t * PP + p] = yi;
        ix[t * PP + p] = xi;
    }
}

// ---------------------------------------------------------------------------
// Kernel 3: ROI pool mean -> hid (f16) [B*N, C]   (feature is L2-resident)
// ---------------------------------------------------------------------------
__global__ void k_hid(const float* __restrict__ feature,    // [B,C,H,W]
                      const int* __restrict__ iy,           // [B*N,7]
                      const int* __restrict__ ix,           // [B*N,7]
                      _Float16* __restrict__ hid) {         // [B*N, C]
    const int total = Bb * Nn * Cc;
    for (int idx = blockIdx.x * blockDim.x + threadIdx.x; idx < total;
         idx += gridDim.x * blockDim.x) {
        const int c  = idx % Cc;
        const int bn = idx / Cc;
        const int b  = bn / Nn;
        int ly[PP], lx[PP];
        #pragma unroll
        for (int p = 0; p < PP; ++p) { ly[p] = iy[bn * PP + p]; lx[p] = ix[bn * PP + p]; }
        const float* f = feature + (size_t)(b * Cc + c) * HW;
        float acc = 0.0f;
        #pragma unroll
        for (int py = 0; py < PP; ++py) {
            const float* row = f + ly[py] * Ww;
            #pragma unroll
            for (int px = 0; px < PP; ++px) acc += row[lx[px]];
        }
        hid[(size_t)bn * Cc + c] = (_Float16)(acc * (1.0f / (PP * PP)));
    }
}

// ---------------------------------------------------------------------------
// Kernel 4: pack W_cls [C,K] f32 into per-lane WMMA B-operand tiles (f16).
// Layout: wpack[((nt*KTILES + kt)*32 + lane)*16 + j] = W[kt*32 + 16*(lane>>4) + j][nt*16 + (lane&15)]
// ---------------------------------------------------------------------------
__global__ void k_pack_w(const float* __restrict__ Wc,      // [C,K]
                         _Float16* __restrict__ wpack) {
    const int total = NTILES * KTILES * 32 * 16;
    for (int idx = blockIdx.x * blockDim.x + threadIdx.x; idx < total;
         idx += gridDim.x * blockDim.x) {
        const int j    = idx & 15;
        const int lane = (idx >> 4) & 31;
        const int kt   = (idx >> 9) % KTILES;
        const int nt   = idx / (16 * 32 * KTILES);
        const int k = kt * 32 + ((lane >> 4) << 4) + j;
        const int n = nt * 16 + (lane & 15);
        wpack[idx] = (_Float16)Wc[(size_t)k * Kk + n];
    }
}

// ---------------------------------------------------------------------------
// Kernel 5: WMMA GEMM: logits[512,80] = hid x W + b. One wave per 16x16 tile.
// ---------------------------------------------------------------------------
__global__ void k_gemm_wmma(const _Float16* __restrict__ hid,   // [512, 2304]
                            const _Float16* __restrict__ wpack, // packed B tiles
                            const float* __restrict__ bias,     // [80]
                            float* __restrict__ logits) {       // [512, 80]
    const int tile = blockIdx.x;           // 0..159
    const int mt   = tile / NTILES;
    const int nt   = tile % NTILES;
    const int lane = threadIdx.x;          // 0..31 (wave32)

    const int row   = mt * 16 + (lane & 15);
    const int khalf = (lane >> 4) << 4;    // 0 or 16

    const _Float16* aptr = hid + (size_t)row * Cc + khalf;
    const _Float16* bptr = wpack + ((size_t)(nt * KTILES) * 32 + lane) * 16;

    v8f acc = {};
    #pragma unroll 4
    for (int kt = 0; kt < KTILES; ++kt) {
        v16h a = *reinterpret_cast<const v16h*>(aptr + (size_t)kt * 32);
        v16h bmat = *reinterpret_cast<const v16h*>(bptr + (size_t)kt * 32 * 16);
        acc = __builtin_amdgcn_wmma_f32_16x16x32_f16(
            /*neg_a=*/false, a, /*neg_b=*/false, bmat,
            /*c_mod=*/(short)0, acc, /*reuse_a=*/false, /*reuse_b=*/false);
    }

    const int ncol = nt * 16 + (lane & 15);
    const float bv = bias[ncol];
    #pragma unroll
    for (int r = 0; r < 8; ++r) {
        const int mrow = mt * 16 + r + ((lane >> 4) << 3);   // lanes 16-31 -> M+8
        logits[(size_t)mrow * Kk + ncol] = acc[r] + bv;
    }
}

// ---------------------------------------------------------------------------
// Kernel 6: multilabel soft-margin loss, fg-masked mean over proposals -> [B]
// ---------------------------------------------------------------------------
__device__ __forceinline__ float log_sigmoid(float x) {
    return fminf(x, 0.0f) - log1pf(expf(-fabsf(x)));
}

__global__ void k_loss(const float* __restrict__ logits,    // [B*N, K]
                       const float* __restrict__ gtc,       // [B,M,K]
                       const int* __restrict__ assign,      // [B*N]
                       const float* __restrict__ fg,        // [B*N]
                       float* __restrict__ out) {           // [B]
    __shared__ float s_num[Nn];
    __shared__ float s_cnt[Nn];
    const int b = blockIdx.x;
    const int n = threadIdx.x;
    const int bn = b * Nn + n;

    const int a = assign[bn];
    const float* lab = gtc + ((size_t)(b * Mm + a)) * Kk;
    const float* lg  = logits + (size_t)bn * Kk;
    float loss = 0.0f;
    #pragma unroll 8
    for (int k = 0; k < Kk; ++k) {
        const float x = lg[k];
        const float y = lab[k];
        loss -= y * log_sigmoid(x) + (1.0f - y) * log_sigmoid(-x);
    }
    const float per = loss * (1.0f / Kk);
    const float f = fg[bn];
    s_num[n] = per * f;
    s_cnt[n] = f;
    __syncthreads();
    for (int s = Nn / 2; s > 0; s >>= 1) {
        if (n < s) { s_num[n] += s_num[n + s]; s_cnt[n] += s_cnt[n + s]; }
        __syncthreads();
    }
    if (n == 0) {
        const float cnt = s_cnt[0];
        out[b] = (cnt > 0.0f) ? (s_num[0] / fmaxf(cnt, 1.0f)) : 0.0f;
    }
}

// ---------------------------------------------------------------------------
// Host launcher
// ---------------------------------------------------------------------------
extern "C" void kernel_launch(void* const* d_in, const int* in_sizes, int n_in,
                              void* d_out, int out_size, void* d_ws, size_t ws_size,
                              hipStream_t stream) {
    const float* fast  = (const float*)d_in[0];
    const float* slow  = (const float*)d_in[1];
    const float* props = (const float*)d_in[2];
    const float* gt    = (const float*)d_in[3];
    const float* gtc   = (const float*)d_in[4];
    const float* Wc    = (const float*)d_in[5];
    const float* bias  = (const float*)d_in[6];
    float* out = (float*)d_out;

    // Workspace layout (all offsets multiples of 256 bytes)
    char* ws = (char*)d_ws;
    float*    feature = (float*)(ws + 0);                        // 7,225,344 B
    _Float16* hid     = (_Float16*)(ws + 7225344);               // 2,359,296 B
    _Float16* wpack   = (_Float16*)(ws + 9584640);               //   368,640 B
    float*    logits  = (float*)(ws + 9953280);                  //   163,840 B
    int*      iy      = (int*)(ws + 10117120);                   //    14,336 B
    int*      ix      = (int*)(ws + 10131456);                   //    14,336 B
    int*      assign  = (int*)(ws + 10145792);                   //     2,048 B
    float*    fg      = (float*)(ws + 10147840);                 //     2,048 B

    // 1. temporal mean + concat (bandwidth-bound: 77 MB read)
    {
        const int total = Bb * Cc * HW;           // 1,806,336
        const int blk = 256;
        const int grd = (total + blk - 1) / blk;
        k_temporal_mean<<<grd, blk, 0, stream>>>(fast, slow, feature);
    }
    // 2. IoU assignment + fg + grid indices (single block for global any())
    k_assign<<<1, Bb * Nn, 0, stream>>>(props, gt, assign, fg, iy, ix);
    // 3. ROI pool mean -> f16 hid
    {
        const int total = Bb * Nn * Cc;           // 1,179,648
        const int blk = 256;
        const int grd = (total + blk - 1) / blk;
        k_hid<<<grd, blk, 0, stream>>>(feature, iy, ix, hid);
    }
    // 4. pack W for WMMA B-operand
    {
        const int total = NTILES * KTILES * 32 * 16;  // 184,320
        const int blk = 256;
        const int grd = (total + blk - 1) / blk;
        k_pack_w<<<grd, blk, 0, stream>>>(Wc, wpack);
    }
    // 5. WMMA GEMM (one wave32 per 16x16 tile; 160 tiles)
    k_gemm_wmma<<<MTILES * NTILES, 32, 0, stream>>>(hid, wpack, bias, logits);
    // 6. loss reduction -> [B]
    k_loss<<<Bb, Nn, 0, stream>>>(logits, gtc, assign, fg, out);
}
